// SwinBlock_12128987644018
// MI455X (gfx1250) — matrix operand order repaired and checked by reading, not compiled, not run on hardware
//
#include <hip/hip_runtime.h>

typedef __bf16 bf16;
typedef bf16  v16bf __attribute__((ext_vector_type(16)));
typedef float v8f   __attribute__((ext_vector_type(8)));

#define WS_   8
#define SS_   4
#define NH_   3
#define C_    96
#define DFF_  384
#define HGT   256
#define NTOK  64
#define QKVC  288

__device__ __forceinline__ int lane_id() { return threadIdx.x & 31; }

// A fragment: 16x32 bf16, element (m,k) at p[m*ld + k]
__device__ __forceinline__ v16bf frag_a(const bf16* p, int ld) {
  const int lane = lane_id();
  const int r = lane & 15, hf = lane >> 4;
  const bf16* row = p + r * ld;
  v16bf a;
#pragma unroll
  for (int j = 0; j < 8; ++j) {
    const int k = 2 * j + hf * 16;
    a[2 * j]     = row[k];
    a[2 * j + 1] = row[k + 1];
  }
  return a;
}

// B fragment: 32x16, B[k][n] = W[n][k] with W row-major ld (k contiguous)
__device__ __forceinline__ v16bf frag_b_rowmajorT(const bf16* p, int ld) {
  const int lane = lane_id();
  const int r = lane & 15, hf = lane >> 4;
  const bf16* row = p + r * ld;
  v16bf b;
#pragma unroll
  for (int j = 0; j < 8; ++j) {
    const int k = 2 * j + hf * 16;
    b[2 * j]     = row[k];
    b[2 * j + 1] = row[k + 1];
  }
  return b;
}

// B fragment: element (k,n) at p[k*ldk + n]  (k strided)
__device__ __forceinline__ v16bf frag_b_strided(const bf16* p, int ldk) {
  const int lane = lane_id();
  const int r = lane & 15, hf = lane >> 4;
  v16bf b;
#pragma unroll
  for (int j = 0; j < 8; ++j) {
    const int k = 2 * j + hf * 16;
    b[2 * j]     = p[k * ldk + r];
    b[2 * j + 1] = p[(k + 1) * ldk + r];
  }
  return b;
}

__device__ __forceinline__ v8f wmma_bf(v16bf a, v16bf b, v8f c) {
  return __builtin_amdgcn_wmma_f32_16x16x32_bf16(false, a, false, b, (short)0, c,
                                                 false, false);
}

__device__ __forceinline__ int region1d(int p) {
  return (p < HGT - WS_) ? 0 : ((p < HGT - SS_) ? 1 : 2);
}

// ---------------- K0: weights fp32 -> bf16 (L2-resident forever after) ------
__global__ void wconv_kernel(const float* __restrict__ qkvw,
                             const float* __restrict__ projw,
                             const float* __restrict__ m1,
                             const float* __restrict__ m2,
                             bf16* __restrict__ out) {
  int i = blockIdx.x * blockDim.x + threadIdx.x;
  if (i >= 110592) return;
  float v;
  if (i < 27648)       v = qkvw[i];
  else if (i < 36864)  v = projw[i - 27648];
  else if (i < 73728)  v = m1[i - 36864];
  else                 v = m2[i - 73728];
  out[i] = (bf16)v;
}

// ---------------- K1: LN1 + shift + window attn + proj + residual ----------
// one workgroup (4 waves) per 8x8 window; writes x1 into d_out
__global__ void __launch_bounds__(128) swin_attn_kernel(
    const float* __restrict__ x,
    const float* __restrict__ n1w, const float* __restrict__ n1b,
    const bf16*  __restrict__ Wqkv, const float* __restrict__ bqkv,
    const float* __restrict__ rpb,
    const bf16*  __restrict__ Wproj, const float* __restrict__ bproj,
    float* __restrict__ x1out) {
  extern __shared__ char smem[];
  float* sX   = (float*)(smem);            // 64x96  f32   (24576 B)
  bf16*  sXN  = (bf16*)(smem + 24576);     // 64x96  bf16  (12288 B)  reused as sO
  bf16*  sO   = sXN;
  bf16*  sQKV = (bf16*)(smem + 36864);     // 64x288 bf16  (36864 B)
  bf16*  sP   = (bf16*)(smem + 73728);     // 3x64x64 bf16 (24576 B)
  float* sRPB = (float*)(smem + 98304);    // 675 f32      (2700 B)

  const int tid  = threadIdx.x;
  const int wave = tid >> 5;
  const int lane = tid & 31;
  const int r15  = lane & 15, hf = lane >> 4;

  const int  img  = blockIdx.x >> 10;
  const int  wIdx = blockIdx.x & 1023;
  const int  wi   = wIdx >> 5, wj = wIdx & 31;
  const long imgBase = (long)img * (HGT * HGT) * C_;

  if (tid == 0) { __builtin_prefetch(Wqkv, 0, 3); __builtin_prefetch(Wproj, 0, 3); }

  for (int i = tid; i < 675; i += 128) sRPB[i] = rpb[i];

  // gather rolled window: rolled[p] = x[(p+SS) mod H]
  for (int i = tid; i < NTOK * C_; i += 128) {
    int t = i / C_, c = i - t * C_;
    int a = t >> 3, b = t & 7;
    int h = (wi * 8 + a + SS_) & (HGT - 1);
    int w = (wj * 8 + b + SS_) & (HGT - 1);
    sX[i] = x[imgBase + (long)(h * HGT + w) * C_ + c];
  }
  __syncthreads();

  // LayerNorm1 per token -> bf16
  if (tid < NTOK) {
    const float* row = sX + tid * C_;
    float m = 0.f;
    for (int c = 0; c < C_; ++c) m += row[c];
    m *= (1.f / C_);
    float v = 0.f;
    for (int c = 0; c < C_; ++c) { float d = row[c] - m; v += d * d; }
    v *= (1.f / C_);
    float rs = rsqrtf(v + 1e-5f);
    for (int c = 0; c < C_; ++c)
      sXN[tid * C_ + c] = (bf16)((row[c] - m) * rs * n1w[c] + n1b[c]);
  }
  __syncthreads();

  // QKV GEMM: (64x96)x(96->288), scale folded into Q
  {
    const int ms = wave * 16;
    v16bf a0 = frag_a(sXN + ms * C_ + 0,  C_);
    v16bf a1 = frag_a(sXN + ms * C_ + 32, C_);
    v16bf a2 = frag_a(sXN + ms * C_ + 64, C_);
    for (int ct = 0; ct < 18; ++ct) {
      const int nb = ct * 16;
      v8f c = {};
      c = wmma_bf(a0, frag_b_rowmajorT(Wqkv + nb * C_ + 0,  C_), c);
      c = wmma_bf(a1, frag_b_rowmajorT(Wqkv + nb * C_ + 32, C_), c);
      c = wmma_bf(a2, frag_b_rowmajorT(Wqkv + nb * C_ + 64, C_), c);
      const int col = nb + r15;
      const float bb = bqkv[col];
      const float qs = (col < C_) ? 0.17677669529663687f : 1.f; // hd^-0.5 on Q
#pragma unroll
      for (int i = 0; i < 8; ++i) {
        int row = ms + i + 8 * hf;
        sQKV[row * QKVC + col] = (bf16)((c[i] + bb) * qs);
      }
    }
  }
  __syncthreads();

  // scores = QK^T + rel-pos bias + shift mask, softmax -> sP (bf16)
  for (int t = wave; t < 12; t += 4) {
    const int hh = t >> 2, s = t & 3;
    v16bf aq = frag_a(sQKV + (s * 16) * QKVC + hh * 32, QKVC);
    v8f c[4] = {};
#pragma unroll
    for (int ct = 0; ct < 4; ++ct)
      c[ct] = wmma_bf(aq, frag_b_rowmajorT(sQKV + (ct * 16) * QKVC + C_ + hh * 32, QKVC), c[ct]);

#pragma unroll
    for (int i = 0; i < 8; ++i) {
      const int row = s * 16 + i + 8 * hf;      // token n
      const int a1 = row >> 3, b1 = row & 7;
      const int idn = 3 * region1d(wi * 8 + a1) + region1d(wj * 8 + b1);
      float v[4];
#pragma unroll
      for (int ct = 0; ct < 4; ++ct) {
        const int col = ct * 16 + r15;          // token m
        const int a2 = col >> 3, b2 = col & 7;
        const int idm = 3 * region1d(wi * 8 + a2) + region1d(wj * 8 + b2);
        const int ridx = (a1 - a2 + 7) * 15 + (b1 - b2 + 7);
        v[ct] = c[ct][i] + sRPB[ridx * NH_ + hh] + ((idn != idm) ? -100.f : 0.f);
      }
      float m = fmaxf(fmaxf(v[0], v[1]), fmaxf(v[2], v[3]));
#pragma unroll
      for (int d = 1; d < 16; d <<= 1) m = fmaxf(m, __shfl_xor(m, d, 16));
      float e[4], ssum = 0.f;
#pragma unroll
      for (int ct = 0; ct < 4; ++ct) { e[ct] = __expf(v[ct] - m); ssum += e[ct]; }
#pragma unroll
      for (int d = 1; d < 16; d <<= 1) ssum += __shfl_xor(ssum, d, 16);
      const float inv = 1.f / ssum;
#pragma unroll
      for (int ct = 0; ct < 4; ++ct)
        sP[hh * 4096 + row * 64 + ct * 16 + r15] = (bf16)(e[ct] * inv);
    }
  }
  __syncthreads();

  // O = P @ V  -> sO (reuses sXN region)
  for (int t = wave; t < 12; t += 4) {
    const int hh = t >> 2, s = t & 3;
    v16bf p0 = frag_a(sP + hh * 4096 + (s * 16) * 64 + 0,  64);
    v16bf p1 = frag_a(sP + hh * 4096 + (s * 16) * 64 + 32, 64);
#pragma unroll
    for (int ct = 0; ct < 2; ++ct) {
      v8f c = {};
      c = wmma_bf(p0, frag_b_strided(sQKV + 0  * QKVC + 192 + hh * 32 + ct * 16, QKVC), c);
      c = wmma_bf(p1, frag_b_strided(sQKV + 32 * QKVC + 192 + hh * 32 + ct * 16, QKVC), c);
      const int col = hh * 32 + ct * 16 + r15;
#pragma unroll
      for (int i = 0; i < 8; ++i) {
        int row = s * 16 + i + 8 * hf;
        sO[row * C_ + col] = (bf16)c[i];
      }
    }
  }
  __syncthreads();

  // proj + roll-back + shortcut residual -> x1 (in d_out)
  {
    const int ms = wave * 16;
    v16bf a0 = frag_a(sO + ms * C_ + 0,  C_);
    v16bf a1 = frag_a(sO + ms * C_ + 32, C_);
    v16bf a2 = frag_a(sO + ms * C_ + 64, C_);
    for (int ct = 0; ct < 6; ++ct) {
      const int nb = ct * 16;
      v8f c = {};
      c = wmma_bf(a0, frag_b_rowmajorT(Wproj + nb * C_ + 0,  C_), c);
      c = wmma_bf(a1, frag_b_rowmajorT(Wproj + nb * C_ + 32, C_), c);
      c = wmma_bf(a2, frag_b_rowmajorT(Wproj + nb * C_ + 64, C_), c);
      const int col = nb + r15;
      const float bb = bproj[col];
#pragma unroll
      for (int i = 0; i < 8; ++i) {
        const int row = ms + i + 8 * hf;
        const int a = row >> 3, b = row & 7;
        const int h = (wi * 8 + a + SS_) & (HGT - 1);
        const int w = (wj * 8 + b + SS_) & (HGT - 1);
        const long g = imgBase + (long)(h * HGT + w) * C_ + col;
        x1out[g] = x[g] + c[i] + bb;
      }
    }
  }
}

// ---------------- K2: LN2 + MLP(96->384 GELU ->96) + residual, in-place ----
__global__ void __launch_bounds__(128) mlp_kernel(
    float* __restrict__ io,    // d_out: reads x1, writes final
    const float* __restrict__ n2w, const float* __restrict__ n2b,
    const bf16*  __restrict__ W1,  const float* __restrict__ b1,
    const bf16*  __restrict__ W2,  const float* __restrict__ b2) {
  extern __shared__ char smem[];
  float* sX  = (float*)(smem);            // 64x96  f32
  bf16*  sXN = (bf16*)(smem + 24576);     // 64x96  bf16
  bf16*  sH  = (bf16*)(smem + 36864);     // 64x384 bf16

  const int tid  = threadIdx.x;
  const int wave = tid >> 5, lane = tid & 31;
  const int r15  = lane & 15, hf = lane >> 4;
  const long base = (long)blockIdx.x * NTOK * C_;

  if (tid == 0) { __builtin_prefetch(W1, 0, 3); __builtin_prefetch(W2, 0, 3); }

  for (int i = tid; i < NTOK * C_; i += 128) sX[i] = io[base + i];
  __syncthreads();

  if (tid < NTOK) {
    const float* row = sX + tid * C_;
    float m = 0.f;
    for (int c = 0; c < C_; ++c) m += row[c];
    m *= (1.f / C_);
    float v = 0.f;
    for (int c = 0; c < C_; ++c) { float d = row[c] - m; v += d * d; }
    v *= (1.f / C_);
    float rs = rsqrtf(v + 1e-5f);
    for (int c = 0; c < C_; ++c)
      sXN[tid * C_ + c] = (bf16)((row[c] - m) * rs * n2w[c] + n2b[c]);
  }
  __syncthreads();

  // GEMM1 (64x96)x(96->384) + exact GELU
  {
    const int ms = wave * 16;
    v16bf a0 = frag_a(sXN + ms * C_ + 0,  C_);
    v16bf a1 = frag_a(sXN + ms * C_ + 32, C_);
    v16bf a2 = frag_a(sXN + ms * C_ + 64, C_);
    for (int ct = 0; ct < 24; ++ct) {
      const int nb = ct * 16;
      v8f c = {};
      c = wmma_bf(a0, frag_b_rowmajorT(W1 + nb * C_ + 0,  C_), c);
      c = wmma_bf(a1, frag_b_rowmajorT(W1 + nb * C_ + 32, C_), c);
      c = wmma_bf(a2, frag_b_rowmajorT(W1 + nb * C_ + 64, C_), c);
      const int col = nb + r15;
      const float bb = b1[col];
#pragma unroll
      for (int i = 0; i < 8; ++i) {
        const int row = ms + i + 8 * hf;
        float u = c[i] + bb;
        sH[row * DFF_ + col] = (bf16)(0.5f * u * (1.f + erff(u * 0.70710678118654752f)));
      }
    }
  }
  __syncthreads();

  // GEMM2 (64x384)x(384->96) + b2 + x1 residual (k-outer, 6 accumulators)
  {
    const int ms = wave * 16;
    v8f acc[6] = {};
#pragma unroll
    for (int kc = 0; kc < 12; ++kc) {
      v16bf a = frag_a(sH + ms * DFF_ + kc * 32, DFF_);
#pragma unroll
      for (int ct = 0; ct < 6; ++ct)
        acc[ct] = wmma_bf(a, frag_b_rowmajorT(W2 + (ct * 16) * DFF_ + kc * 32, DFF_), acc[ct]);
    }
#pragma unroll
    for (int ct = 0; ct < 6; ++ct) {
      const int col = ct * 16 + r15;
      const float bb = b2[col];
#pragma unroll
      for (int i = 0; i < 8; ++i) {
        const int row = ms + i + 8 * hf;
        io[base + row * C_ + col] = sX[row * C_ + col] + acc[ct][i] + bb;
      }
    }
  }
}

extern "C" void kernel_launch(void* const* d_in, const int* in_sizes, int n_in,
                              void* d_out, int out_size, void* d_ws, size_t ws_size,
                              hipStream_t stream) {
  (void)in_sizes; (void)n_in; (void)out_size; (void)ws_size;
  const float* x    = (const float*)d_in[0];
  const float* n1w  = (const float*)d_in[1];
  const float* n1b  = (const float*)d_in[2];
  const float* qkvw = (const float*)d_in[3];
  const float* qkvb = (const float*)d_in[4];
  const float* rpb  = (const float*)d_in[5];
  const float* pw   = (const float*)d_in[6];
  const float* pb   = (const float*)d_in[7];
  const float* n2w  = (const float*)d_in[8];
  const float* n2b  = (const float*)d_in[9];
  const float* w1   = (const float*)d_in[10];
  const float* b1   = (const float*)d_in[11];
  const float* w2   = (const float*)d_in[12];
  const float* b2   = (const float*)d_in[13];
  float* out = (float*)d_out;

  bf16* wbf   = (bf16*)d_ws;
  bf16* Wqkv  = wbf;           // 288*96
  bf16* Wproj = wbf + 27648;   // 96*96
  bf16* W1    = wbf + 36864;   // 384*96
  bf16* W2    = wbf + 73728;   // 96*384

  wconv_kernel<<<(110592 + 255) / 256, 256, 0, stream>>>(qkvw, pw, w1, w2, wbf);
  swin_attn_kernel<<<8192, 128, 101376, stream>>>(x, n1w, n1b, Wqkv, qkvb, rpb,
                                                  Wproj, pb, out);
  mlp_kernel<<<8192, 128, 86016, stream>>>(out, n2w, n2b, W1, b1, W2, b2);
}
